// MPNN_23398981828719
// MI455X (gfx1250) — matrix operand order repaired and checked by reading, not compiled, not run on hardware
//
#include <hip/hip_runtime.h>
#include <hip/hip_bf16.h>
#include <math.h>

// ---------------------------------------------------------------------------
// Problem constants (from reference setup_inputs)
// ---------------------------------------------------------------------------
#define Bb   32
#define Nn   512
#define Din  64
#define Hh   128
#define E1   4
#define Dout 16
#define BN   (Bb * Nn)          // 16384
#define NUM_PROP 5
#define NUM_S2V  3

typedef __attribute__((ext_vector_type(16))) __bf16 v16bf;
typedef __attribute__((ext_vector_type(8)))  __bf16 v8bf;
typedef __attribute__((ext_vector_type(8)))  float  v8f;

// ---------------------------------------------------------------------------
// Generic batched bf16 GEMM:  C[M,N] = A[M,K] * Bt[N,K]^T   (NT form)
// A row-major (lda = K stride), Bt row-major (ldb = K stride).
// Each wave computes a 64(M) x 32(N) tile: 8 v_wmma_f32_16x16x32_bf16 per
// 32-K step from 6 fragment loads.  K is manually unrolled by 64 with two
// fragment buffer sets (ping-pong, no rotation copies) so the loads of one
// half-step overlap the WMMA group of the other.  Requires K % 64 == 0.
// blockDim = 128 (4 waves -> 128 N cols).  __launch_bounds__(128,1) gives
// the backend a full VGPR budget (~180/wave needed) so nothing spills.
// Batch z: operand offsets = (z/divE)*s1 + (z%divE)*s2  (element units).
// out_mode: 0 = f32 row-major, 1 = bf16 row-major, 2 = bf16 transposed
//           (C[col*ldc + row], used to emit msg^T for the agg GEMM).
// ---------------------------------------------------------------------------
__global__ void __launch_bounds__(128, 1)
gemm_bf16_nt(const __bf16* __restrict__ A,
             const __bf16* __restrict__ Bt,
             void* __restrict__ C,
             int M, int N, int K,
             int lda, int ldb, int ldc,
             int divE,
             long long sA1, long long sA2,
             long long sB1, long long sB2,
             long long sC1, long long sC2,
             int out_mode)
{
    const int z  = blockIdx.z;
    const long long z1 = z / divE, z2 = z % divE;
    A  += (size_t)(z1 * sA1 + z2 * sA2);
    Bt += (size_t)(z1 * sB1 + z2 * sB2);
    const size_t coff = (size_t)(z1 * sC1 + z2 * sC2);
    float*  Cf = (float*)C  + coff;
    __bf16* Ch = (__bf16*)C + coff;

    const int lane  = threadIdx.x & 31;
    const int wave  = threadIdx.x >> 5;
    const int nBase = (blockIdx.y * 4 + wave) * 32;     // 32 N-cols per wave
    if (nBase >= N) return;
    const int mBase = blockIdx.x * 64;

    const int lm = lane & 15;        // row (A) / col (B,C) within tile
    const int hi = lane >> 4;        // half-wave select
    const int kA = hi * 16;          // A fragment K offset (0 or 16)
    const int kB = hi * 16;          // B fragment K offset (0 or 16)

    const __bf16* aBase = A + (size_t)(mBase + lm) * lda + kA;
    const __bf16* b0p   = Bt + (size_t)(nBase + lm) * ldb + kB;
    const __bf16* b1p   = b0p + (size_t)16 * ldb;
    const size_t aRowStride = (size_t)16 * lda;

    v8f acc[4][2] = {};

    auto loadA = [&](int k, v16bf a[4]) {
#pragma unroll
        for (int t = 0; t < 4; ++t) {
            const __bf16* p = aBase + (size_t)t * aRowStride + k;
            // A fragment contiguous (matches B): K = kA..kA+15
            v8bf lo  = *(const v8bf*)(p);
            v8bf hi8 = *(const v8bf*)(p + 8);
#pragma unroll
            for (int i = 0; i < 8; ++i) { a[t][i] = lo[i]; a[t][8 + i] = hi8[i]; }
        }
    };
    auto loadB = [&](int k, v16bf b[2]) {
        b[0] = *(const v16bf*)(b0p + k);
        b[1] = *(const v16bf*)(b1p + k);
    };
    auto wmmaGroup = [&](const v16bf a[4], const v16bf b[2]) {
#pragma unroll
        for (int t = 0; t < 4; ++t) {
            acc[t][0] = __builtin_amdgcn_wmma_f32_16x16x32_bf16(
                false, a[t], false, b[0], (short)0, acc[t][0], false, false);
            acc[t][1] = __builtin_amdgcn_wmma_f32_16x16x32_bf16(
                false, a[t], false, b[1], (short)0, acc[t][1], false, false);
        }
    };

    // ping-pong pipeline over 64-wide K super-steps (K % 64 == 0)
    v16bf a0[4], a1[4], b0[2], b1[2];
    loadA(0, a0); loadB(0, b0);
    for (int k = 0; k < K; k += 64) {
        loadA(k + 32, a1); loadB(k + 32, b1);
        __builtin_prefetch(aBase + k + 256, 0, 1);     // global_prefetch_b8
        wmmaGroup(a0, b0);
        if (k + 64 < K) { loadA(k + 64, a0); loadB(k + 64, b0); }
        wmmaGroup(a1, b1);
    }

#pragma unroll
    for (int u = 0; u < 2; ++u) {
        const int col = nBase + u * 16 + lm;
#pragma unroll
        for (int t = 0; t < 4; ++t) {
#pragma unroll
            for (int v = 0; v < 8; ++v) {
                const int row = mBase + t * 16 + v + hi * 8;
                const float val = acc[t][u][v];
                if (out_mode == 0)      Cf[(size_t)row * ldc + col] = val;
                else if (out_mode == 1) Ch[(size_t)row * ldc + col] = (__bf16)val;
                else                    Ch[(size_t)col * ldc + row] = (__bf16)val;
            }
        }
    }
}

// ---------------------------------------------------------------------------
// Binarize adjacency L[B,N,N,E1] (f32) -> adj bf16 [B,E1,N,N]
// e fastest-varying per thread => fully coalesced reads of L.
// ---------------------------------------------------------------------------
__global__ void binarize_kernel(const float* __restrict__ L,
                                __bf16* __restrict__ adj)
{
    const long long idx = (long long)blockIdx.x * blockDim.x + threadIdx.x;
    const long long total = (long long)Bb * Nn * Nn * E1;
    if (idx >= total) return;
    const int e = idx & (E1 - 1);
    const int n = (int)(idx >> 2) & (Nn - 1);
    const int m = (int)(idx >> 11) & (Nn - 1);
    const int b = (int)(idx >> 20);
    const float v = L[idx];
    adj[((((size_t)b * E1 + e) * Nn + m) * Nn) + n] =
        (__bf16)((v != 0.0f) ? 1.0f : 0.0f);
}

// ---------------------------------------------------------------------------
// state0[b,n,h] = node_emb[node_feat[b,n]] . W_in[h,:] + b_in[h]
// ---------------------------------------------------------------------------
__global__ void embed_kernel(const int* __restrict__ node_feat,
                             const float* __restrict__ node_emb,
                             const float* __restrict__ W_in,
                             const float* __restrict__ b_in,
                             float* __restrict__ stateF,
                             __bf16* __restrict__ stateB)
{
    const int idx = blockIdx.x * blockDim.x + threadIdx.x;   // BN*H
    if (idx >= BN * Hh) return;
    const int node = idx >> 7;
    const int h    = idx & (Hh - 1);
    const float* emb = node_emb + (size_t)node_feat[node] * Din;
    const float* w   = W_in + (size_t)h * Din;
    float acc = b_in[h];
#pragma unroll 4
    for (int i = 0; i < Din; ++i) acc += emb[i] * w[i];
    stateF[idx] = acc;
    stateB[idx] = (__bf16)acc;
}

// ---------------------------------------------------------------------------
// f32 -> bf16 elementwise
// ---------------------------------------------------------------------------
__global__ void cvt_bf16_kernel(const float* __restrict__ src,
                                __bf16* __restrict__ dst, int n)
{
    const int idx = blockIdx.x * blockDim.x + threadIdx.x;
    if (idx < n) dst[idx] = (__bf16)src[idx];
}

// ---------------------------------------------------------------------------
// EopT[e][kout][h] = edge_emb[e][h*H + kout]  (transpose to NT "Bt" layout)
// ---------------------------------------------------------------------------
__global__ void eopT_kernel(const float* __restrict__ edge_emb,
                            __bf16* __restrict__ eopT)
{
    const int idx = blockIdx.x * blockDim.x + threadIdx.x;   // E1*H*H
    if (idx >= E1 * Hh * Hh) return;
    const int h    = idx & (Hh - 1);
    const int kout = (idx >> 7) & (Hh - 1);
    const int e    = idx >> 14;
    eopT[idx] = (__bf16)edge_emb[(size_t)e * Hh * Hh + (size_t)h * Hh + kout];
}

// ---------------------------------------------------------------------------
// GRU elementwise update (PyTorch gate order r,z,n), reads gi/gh [BN,3H]
// ---------------------------------------------------------------------------
__global__ void gru_gate_kernel(const float* __restrict__ gi,
                                const float* __restrict__ gh,
                                const float* __restrict__ bih,
                                const float* __restrict__ bhh,
                                float* __restrict__ stateF,
                                __bf16* __restrict__ stateB)
{
    const int idx = blockIdx.x * blockDim.x + threadIdx.x;   // BN*H
    if (idx >= BN * Hh) return;
    const int node = idx >> 7;
    const int h    = idx & (Hh - 1);
    const float* gir = gi + (size_t)node * (3 * Hh);
    const float* ghr = gh + (size_t)node * (3 * Hh);
    const float ir  = gir[h]            + bih[h];
    const float iz  = gir[Hh + h]       + bih[Hh + h];
    const float in_ = gir[2 * Hh + h]   + bih[2 * Hh + h];
    const float hr  = ghr[h]            + bhh[h];
    const float hz  = ghr[Hh + h]       + bhh[Hh + h];
    const float hn  = ghr[2 * Hh + h]   + bhh[2 * Hh + h];
    const float r = 1.0f / (1.0f + expf(-(ir + hr)));
    const float z = 1.0f / (1.0f + expf(-(iz + hz)));
    const float n = tanhf(in_ + r * hn);
    const float hp = stateF[idx];
    const float s = (1.0f - z) * n + z * hp;
    stateF[idx] = s;
    stateB[idx] = (__bf16)s;
}

// ---------------------------------------------------------------------------
// Set2Vec readout + output linear. One 256-thread block per graph.
// ---------------------------------------------------------------------------
__global__ void set2vec_out_kernel(const float* __restrict__ state,
                                   const float* __restrict__ wih,   // [4H, 2H]
                                   const float* __restrict__ whh,   // [4H, H]
                                   const float* __restrict__ bih,
                                   const float* __restrict__ bhh,
                                   const float* __restrict__ Wout,  // [Dout, 2H]
                                   const float* __restrict__ bout,
                                   float* __restrict__ out)
{
    __shared__ float h[Hh], c[Hh], q[2 * Hh], g[4 * Hh];
    __shared__ float logits[Nn];
    __shared__ float sred[256];

    const int b = blockIdx.x;
    const int t = threadIdx.x;               // 256 threads
    const float* m = state + (size_t)b * Nn * Hh;

    if (t < Hh) { h[t] = 0.0f; c[t] = 0.0f; }
    q[t] = 0.0f;
    __syncthreads();

    for (int step = 0; step < NUM_S2V; ++step) {
        // g[j] = bih[j] + bhh[j] + q.wih[j,:] + h.whh[j,:]   (j = t, t+256)
        for (int jj = 0; jj < 2; ++jj) {
            const int j = t + jj * 256;
            float acc = bih[j] + bhh[j];
            const float* wi = wih + (size_t)j * (2 * Hh);
            for (int i = 0; i < 2 * Hh; ++i) acc += q[i] * wi[i];
            const float* wh = whh + (size_t)j * Hh;
            for (int i = 0; i < Hh; ++i) acc += h[i] * wh[i];
            g[j] = acc;
        }
        __syncthreads();
        if (t < Hh) {
            const float ig = 1.0f / (1.0f + expf(-g[t]));
            const float fg = 1.0f / (1.0f + expf(-g[Hh + t]));
            const float gg = tanhf(g[2 * Hh + t]);
            const float og = 1.0f / (1.0f + expf(-g[3 * Hh + t]));
            const float c2 = fg * c[t] + ig * gg;
            c[t] = c2;
            h[t] = og * tanhf(c2);
        }
        __syncthreads();
        // attention logits over the node set
        for (int n = t; n < Nn; n += 256) {
            const float* mr = m + (size_t)n * Hh;
            float acc = 0.0f;
            for (int i = 0; i < Hh; ++i) acc += mr[i] * h[i];
            logits[n] = acc;
        }
        __syncthreads();
        // softmax: max reduce
        float lm = fmaxf(logits[t], logits[t + 256]);
        sred[t] = lm;
        __syncthreads();
        for (int s = 128; s > 0; s >>= 1) {
            if (t < s) sred[t] = fmaxf(sred[t], sred[t + s]);
            __syncthreads();
        }
        const float gmax = sred[0];
        __syncthreads();
        // exp + sum reduce
        float e0 = expf(logits[t] - gmax);
        float e1 = expf(logits[t + 256] - gmax);
        logits[t] = e0; logits[t + 256] = e1;
        sred[t] = e0 + e1;
        __syncthreads();
        for (int s = 128; s > 0; s >>= 1) {
            if (t < s) sred[t] += sred[t + s];
            __syncthreads();
        }
        const float inv = 1.0f / sred[0];
        __syncthreads();
        // read vector r[k] and new q = [h, r]
        if (t < Hh) {
            float r = 0.0f;
            for (int n = 0; n < Nn; ++n) r += logits[n] * m[(size_t)n * Hh + t];
            q[t] = h[t];
            q[Hh + t] = r * inv;
        }
        __syncthreads();
    }

    if (t < Dout) {
        const float* wo = Wout + (size_t)t * (2 * Hh);
        float acc = bout[t];
        for (int j = 0; j < 2 * Hh; ++j) acc += q[j] * wo[j];
        out[(size_t)b * Dout + t] = acc;
    }
}

// ---------------------------------------------------------------------------
// Host orchestration
// ---------------------------------------------------------------------------
extern "C" void kernel_launch(void* const* d_in, const int* in_sizes, int n_in,
                              void* d_out, int out_size, void* d_ws, size_t ws_size,
                              hipStream_t stream)
{
    (void)in_sizes; (void)n_in; (void)out_size; (void)ws_size;

    const int*   node_feat = (const int*)  d_in[0];
    const float* L         = (const float*)d_in[1];
    const float* node_emb  = (const float*)d_in[2];
    const float* W_in      = (const float*)d_in[3];
    const float* b_in      = (const float*)d_in[4];
    const float* edge_emb  = (const float*)d_in[5];
    const float* gru_wih   = (const float*)d_in[6];
    const float* gru_whh   = (const float*)d_in[7];
    const float* gru_bih   = (const float*)d_in[8];
    const float* gru_bhh   = (const float*)d_in[9];
    const float* lstm_wih  = (const float*)d_in[10];
    const float* lstm_whh  = (const float*)d_in[11];
    const float* lstm_bih  = (const float*)d_in[12];
    const float* lstm_bhh  = (const float*)d_in[13];
    const float* W_out     = (const float*)d_in[14];
    const float* b_out     = (const float*)d_in[15];
    float* out = (float*)d_out;

    // workspace carve-up (256B aligned)
    char* ws = (char*)d_ws;
    auto alloc = [&](size_t bytes) -> char* {
        char* p = ws; ws += (bytes + 255) & ~(size_t)255; return p;
    };
    __bf16* adj    = (__bf16*)alloc((size_t)Bb * E1 * Nn * Nn * 2);   // 64 MB (L2-resident)
    float*  stateF = (float*) alloc((size_t)BN * Hh * 4);
    __bf16* stateB = (__bf16*)alloc((size_t)BN * Hh * 2);
    __bf16* msgT   = (__bf16*)alloc((size_t)Bb * E1 * Hh * Nn * 2);   // [b,e,kout,n]
    __bf16* aggB   = (__bf16*)alloc((size_t)BN * (E1 * Hh) * 2);      // msg_cat bf16
    float*  giF    = (float*) alloc((size_t)BN * 3 * Hh * 4);
    float*  ghF    = (float*) alloc((size_t)BN * 3 * Hh * 4);
    __bf16* eopT   = (__bf16*)alloc((size_t)E1 * Hh * Hh * 2);
    __bf16* wihB   = (__bf16*)alloc((size_t)3 * Hh * E1 * Hh * 2);
    __bf16* whhB   = (__bf16*)alloc((size_t)3 * Hh * Hh * 2);

    // --- prep ---
    {
        const long long tot = (long long)Bb * Nn * Nn * E1;
        binarize_kernel<<<dim3((unsigned)((tot + 255) / 256)), dim3(256), 0, stream>>>(L, adj);
    }
    embed_kernel<<<dim3((BN * Hh + 255) / 256), dim3(256), 0, stream>>>(
        node_feat, node_emb, W_in, b_in, stateF, stateB);
    cvt_bf16_kernel<<<dim3((3 * Hh * E1 * Hh + 255) / 256), dim3(256), 0, stream>>>(
        gru_wih, wihB, 3 * Hh * E1 * Hh);
    cvt_bf16_kernel<<<dim3((3 * Hh * Hh + 255) / 256), dim3(256), 0, stream>>>(
        gru_whh, whhB, 3 * Hh * Hh);
    eopT_kernel<<<dim3((E1 * Hh * Hh + 255) / 256), dim3(256), 0, stream>>>(edge_emb, eopT);

    // --- propagation rounds ---
    for (int p = 0; p < NUM_PROP; ++p) {
        // msg^T[b,e,k,n] = (state[b] @ Eop[e])^T       (batched over z = b*E1+e)
        gemm_bf16_nt<<<dim3(Nn / 64, 1, Bb * E1), dim3(128), 0, stream>>>(
            stateB, eopT, msgT,
            Nn, Hh, Hh, Hh, Hh, /*ldc=*/Nn,
            E1,
            /*A*/ (long long)Nn * Hh, 0,
            /*B*/ 0, (long long)Hh * Hh,
            /*C*/ (long long)E1 * Hh * Nn, (long long)Hh * Nn,
            /*out_mode=*/2);

        // agg[b,m,e,k] = Adj[b,e] @ msg[b,e]           (batched over z = b*E1+e)
        gemm_bf16_nt<<<dim3(Nn / 64, 1, Bb * E1), dim3(128), 0, stream>>>(
            adj, msgT, aggB,
            Nn, Hh, Nn, Nn, Nn, /*ldc=*/E1 * Hh,
            E1,
            /*A*/ (long long)E1 * Nn * Nn, (long long)Nn * Nn,
            /*B*/ (long long)E1 * Hh * Nn, (long long)Hh * Nn,
            /*C*/ (long long)Nn * E1 * Hh, (long long)Hh,
            /*out_mode=*/1);

        // gi = msg_cat @ gru_wih^T
        gemm_bf16_nt<<<dim3(BN / 64, (3 * Hh) / 128, 1), dim3(128), 0, stream>>>(
            aggB, wihB, giF,
            BN, 3 * Hh, E1 * Hh, E1 * Hh, E1 * Hh, 3 * Hh,
            1, 0, 0, 0, 0, 0, 0, /*out_mode=*/0);

        // gh = state @ gru_whh^T
        gemm_bf16_nt<<<dim3(BN / 64, (3 * Hh) / 128, 1), dim3(128), 0, stream>>>(
            stateB, whhB, ghF,
            BN, 3 * Hh, Hh, Hh, Hh, 3 * Hh,
            1, 0, 0, 0, 0, 0, 0, /*out_mode=*/0);

        // GRU elementwise update
        gru_gate_kernel<<<dim3((BN * Hh + 255) / 256), dim3(256), 0, stream>>>(
            giF, ghF, gru_bih, gru_bhh, stateF, stateB);
    }

    // --- Set2Vec readout + output linear ---
    set2vec_out_kernel<<<dim3(Bb), dim3(256), 0, stream>>>(
        stateF, lstm_wih, lstm_whh, lstm_bih, lstm_bhh, W_out, b_out, out);
}
